// ImprovedGCN_9302899163452
// MI455X (gfx1250) — compile-verified, matrix operand order checked by reading, not run on hardware
//
#include <hip/hip_runtime.h>
#include <hip/hip_bf16.h>
#include <math.h>

typedef __bf16 bf16_t;
typedef __attribute__((ext_vector_type(16))) __bf16 v16bf;
typedef __attribute__((ext_vector_type(8)))  __bf16 v8bf;
typedef __attribute__((ext_vector_type(8)))  float  v8f;
typedef __attribute__((ext_vector_type(4)))  unsigned int u32x4;
typedef __attribute__((ext_vector_type(8)))  int i32x8;
typedef __attribute__((ext_vector_type(4)))  int i32x4;

#if defined(__has_builtin)
#  if __has_builtin(__builtin_amdgcn_tensor_load_to_lds) && \
      __has_builtin(__builtin_amdgcn_s_wait_tensorcnt)
#    define HAVE_TDM 1
#  endif
#endif
#ifndef HAVE_TDM
#  define HAVE_TDM 0
#endif

// ---- helpers ---------------------------------------------------------------

static __device__ __forceinline__ bf16_t f2bf(float f) {
    union { float f; unsigned u; } c; c.f = f;
    unsigned r = c.u + 0x7FFFu + ((c.u >> 16) & 1u);   // round-to-nearest-even
    union { unsigned short s; bf16_t b; } o; o.s = (unsigned short)(r >> 16);
    return o.b;
}

static __device__ __forceinline__ void atomic_add_f32(float* p, float v) {
    __hip_atomic_fetch_add(p, v, __ATOMIC_RELAXED, __HIP_MEMORY_SCOPE_AGENT);
}

static __device__ __forceinline__ v8f wmma_bf16(v16bf a, v16bf b, v8f c) {
    return __builtin_amdgcn_wmma_f32_16x16x32_bf16(false, a, false, b,
                                                   (short)0, c, false, false);
}

#if HAVE_TDM
// Issue a TDM DMA: 16 rows x K bf16 tile (row-major, stride K) from global to
// LDS at byte offset lds_off, inserting 16B of LDS padding after every 256B
// row chunk (pad_interval=64 DWORDs, pad_amount=4 DWORDs) => padded row
// stride = (K+8) bf16 elements, matching the compute-side layout.
static __device__ __forceinline__ void tdm_load_tile(unsigned lds_off,
                                                     const bf16_t* gsrc,
                                                     int K_elems) {
    unsigned long long ga = (unsigned long long)(uintptr_t)gsrc;
    u32x4 g0;
    g0.x = 1u;                                              // count=1 (valid D#)
    g0.y = lds_off;                                         // lds_addr
    g0.z = (unsigned)ga;                                    // global_addr[31:0]
    g0.w = (unsigned)((ga >> 32) & 0x01FFFFFFu) | (2u << 30); // addr[56:32]|type=2

    i32x8 g1;
    unsigned w0 = (1u << 16)     // data_size = 2 bytes
                | (1u << 20)     // pad_enable
                | (5u << 22)     // pad_interval: 64 DWORDs (256B)
                | (3u << 25);    // pad_amount: 4 DWORDs (16B)
    g1[0] = (int)w0;
    g1[1] = (int)((unsigned)(K_elems & 0xFFFF) << 16);      // tensor_dim0[15:0]
    g1[2] = (int)(((unsigned)K_elems >> 16) | (16u << 16)); // td0[31:16] | td1[15:0]=16
    g1[3] = (int)((unsigned)(K_elems & 0xFFFF) << 16);      // td1[31:16]=0 | tile_dim0
    g1[4] = 16;                                             // tile_dim1=16, tile_dim2=0
    g1[5] = K_elems;                                        // tensor_dim0_stride[31:0]
    g1[6] = 0;                                              // stride hi | td1_stride lo
    g1[7] = 0;
    i32x4 gz = {0, 0, 0, 0};
#if __clang_major__ >= 23
    i32x8 gz8 = {0, 0, 0, 0, 0, 0, 0, 0};
    __builtin_amdgcn_tensor_load_to_lds(g0, g1, gz, gz, gz8, 0);
#else
    __builtin_amdgcn_tensor_load_to_lds(g0, g1, gz, gz, 0);
#endif
}
#endif

// ---- prep kernels ----------------------------------------------------------

__global__ void f32_to_bf16_kernel(const float* __restrict__ in,
                                   bf16_t* __restrict__ out, long n) {
    long i = (long)blockIdx.x * blockDim.x + threadIdx.x;
    if (i < n) out[i] = f2bf(in[i]);
}

// Pack W (K x Dn, row-major f32) into bf16 WMMA B-fragments.
// Fragment f = nt*(K/32)+kt holds 32 lanes x 16 elems; lane L (col n = nt*16+(L&15)),
// elem e -> k = kt*32 + 8*(L>>4) + (e<8 ? e : e+8)   [16-bit A/B VGPR layout]
__global__ void pack_w_kernel(const float* __restrict__ W,
                              bf16_t* __restrict__ out, int K, int Dn) {
    int tid = blockIdx.x * blockDim.x + threadIdx.x;
    if (tid >= K * Dn) return;
    int e    = tid & 15;
    int lane = (tid >> 4) & 31;
    int f    = tid >> 9;
    int KT   = K >> 5;
    int kt   = f % KT;
    int nt   = f / KT;
    int n  = nt * 16 + (lane & 15);
    int ks = (lane >> 4) * 8;
    int k  = kt * 32 + ks + (e < 8 ? e : e + 8);
    out[tid] = f2bf(W[(size_t)k * Dn + n]);
}

// ---- dual GEMM: hn = Hb@Wn ; agg = Hb@Wo + b -------------------------------
// Persistent waves, B fragments in registers. A tile staged in LDS:
//  - TDM path: double-buffered tensor_load_to_lds DMA, pipelined one tile ahead
//  - fallback: synchronous coalesced copy
template<int K, int DT>
__global__ __launch_bounds__(DT * 32)
void gemm_dual_kernel(const bf16_t* __restrict__ Hb,
                      const bf16_t* __restrict__ Wn,
                      const bf16_t* __restrict__ Wo,
                      const float*  __restrict__ bias,
                      float* __restrict__ hn,
                      float* __restrict__ agg,
                      int rowTiles) {
    constexpr int KT   = K / 32;
    constexpr int Dn   = DT * 16;
    constexpr int LROW = K + 8;              // padded LDS row (bf16 elems)
    constexpr int NTHR = DT * 32;
    constexpr int NBUF = HAVE_TDM ? 2 : 1;

    __shared__ bf16_t As[NBUF * 16 * LROW];

    const int lane = threadIdx.x & 31;
    const int wave = threadIdx.x >> 5;       // column tile index

    // ---- hoist weights into registers (invariant over row tiles) ----------
    v16bf fn[KT], fo[KT];
    {
        const bf16_t* bn = Wn + ((size_t)wave * KT) * 512 + (size_t)lane * 16;
        const bf16_t* bo = Wo + ((size_t)wave * KT) * 512 + (size_t)lane * 16;
#pragma unroll
        for (int kt = 0; kt < KT; ++kt) {
            fn[kt] = *(const v16bf*)(bn + (size_t)kt * 512);
            fo[kt] = *(const v16bf*)(bo + (size_t)kt * 512);
        }
    }
    const int   col = wave * 16 + (lane & 15);
    const float bv  = bias[col];
    const int   arow = lane & 15;
    const int   ks   = (lane >> 4) * 8;

#if HAVE_TDM
    const unsigned ldsBase = (unsigned)(uintptr_t)&As[0];
    // prologue: kick DMA for this block's first tile
    if (wave == 0 && (int)blockIdx.x < rowTiles)
        tdm_load_tile(ldsBase, Hb + (size_t)blockIdx.x * 16 * K, K);
    int bufi = 0;
#endif

    for (int rt = blockIdx.x; rt < rowTiles; rt += gridDim.x) {
#if HAVE_TDM
        if (wave == 0) __builtin_amdgcn_s_wait_tensorcnt(0);
        __syncthreads();                     // cur buffer ready; prev reads done
        {
            int nrt = rt + (int)gridDim.x;   // pipeline next tile into other buf
            if (wave == 0 && nrt < rowTiles)
                tdm_load_tile(ldsBase + (unsigned)((bufi ^ 1) * 16 * LROW * 2),
                              Hb + (size_t)nrt * 16 * K, K);
        }
        const bf16_t* At = As + bufi * 16 * LROW;
#else
        __syncthreads();                     // As reuse from previous iter
        constexpr int CHUNKS = 16 * K / 8;   // 16-byte copy chunks per tile
#pragma unroll
        for (int ch = threadIdx.x; ch < CHUNKS; ch += NTHR) {
            int r = ch / (K / 8);
            int c = (ch % (K / 8)) * 8;
            *(v8bf*)(As + r * LROW + c) =
                *(const v8bf*)(Hb + (size_t)(rt * 16 + r) * K + c);
        }
        {
            int nrt = rt + (int)gridDim.x;
            if (nrt < rowTiles)
                __builtin_prefetch(Hb + (size_t)nrt * 16 * K +
                                   (size_t)threadIdx.x * (16 * K / NTHR), 0, 0);
        }
        __syncthreads();
        const bf16_t* At = As;
#endif

        v8f cn = {}; v8f co = {};
#pragma unroll
        for (int kt = 0; kt < KT; ++kt) {
            v8bf lo = *(const v8bf*)(At + arow * LROW + ks + kt * 32);
            v8bf hi = *(const v8bf*)(At + arow * LROW + ks + kt * 32 + 16);
            v16bf a = __builtin_shufflevector(lo, hi,
                        0,1,2,3,4,5,6,7,8,9,10,11,12,13,14,15);
            cn = wmma_bf16(a, fn[kt], cn);
            co = wmma_bf16(a, fo[kt], co);
        }

        // C/D layout: VGPR r -> row rt*16 + r + 8*(lane>=16), col = wave*16+(lane&15)
        // single base pointer per matrix -> stores become base + imm offsets
        const int rbase = rt * 16 + (lane >> 4) * 8;
        float* hp = hn  + (size_t)rbase * Dn + col;
        float* ap = agg + (size_t)rbase * Dn + col;
#pragma unroll
        for (int r = 0; r < 8; ++r) {
            hp[(size_t)r * Dn] = cn[r];
            ap[(size_t)r * Dn] = co[r] + bv; // accumulator seeded with own+bias
        }
#if HAVE_TDM
        bufi ^= 1;
#endif
    }
}

// ---- SpMM scatter: agg[rows[e]] += vals[e] * hn[cols[e]] -------------------
// Each lane loads one edge's metadata (coalesced); the wave then processes its
// 32 edges via shuffle broadcast, lane covering Dn/32 contiguous columns.
template<int Dn>
__global__ __launch_bounds__(256)
void spmm_scatter_kernel(const int*   __restrict__ rows,
                         const int*   __restrict__ cols,
                         const float* __restrict__ vals,
                         const float* __restrict__ hn,
                         float* __restrict__ agg,
                         int E) {
    constexpr int PER = Dn / 32;             // floats per lane (4 or 2)
    const int lane = threadIdx.x & 31;
    const long wid = (long)blockIdx.x * (blockDim.x >> 5) + (threadIdx.x >> 5);
    const long base = wid * 32;
    if (base >= E) return;

    const long me = base + lane;
    int   r = 0, c = 0;
    float v = 0.0f;
    if (me < E) { r = rows[me]; c = cols[me]; v = vals[me]; }
    const int cnt = (int)((E - base) < 32 ? (E - base) : 32);

    for (int j = 0; j < cnt; ++j) {
        int   cj = __shfl(c, j, 32);
        int   rj = __shfl(r, j, 32);
        float vj = __shfl(v, j, 32);
        const float* src = hn  + (size_t)cj * Dn + lane * PER;
        float*       dst = agg + (size_t)rj * Dn + lane * PER;
        if constexpr (PER == 4) {
            float4 d = *(const float4*)src;
            atomic_add_f32(dst + 0, vj * d.x);
            atomic_add_f32(dst + 1, vj * d.y);
            atomic_add_f32(dst + 2, vj * d.z);
            atomic_add_f32(dst + 3, vj * d.w);
        } else {
            float2 d = *(const float2*)src;
            atomic_add_f32(dst + 0, vj * d.x);
            atomic_add_f32(dst + 1, vj * d.y);
        }
    }
}

// ---- tanh + downconvert for next layer ------------------------------------
__global__ void tanh_to_bf16_kernel(const float* __restrict__ agg,
                                    bf16_t* __restrict__ outb, long n) {
    long i = (long)blockIdx.x * blockDim.x + threadIdx.x;
    if (i < n) outb[i] = f2bf(tanhf(agg[i]));
}

// ---- host-side orchestration ----------------------------------------------

static constexpr int N_NODES = 100000;
static constexpr int N_EDGES = 1600000;

extern "C" void kernel_launch(void* const* d_in, const int* in_sizes, int n_in,
                              void* d_out, int out_size, void* d_ws, size_t ws_size,
                              hipStream_t stream) {
    const float* x    = (const float*)d_in[0];
    const int*   rows = (const int*)  d_in[1];
    const int*   cols = (const int*)  d_in[2];
    const float* vals = (const float*)d_in[3];
    const float* Wn0  = (const float*)d_in[4];
    const float* Wo0  = (const float*)d_in[5];
    const float* b0   = (const float*)d_in[6];
    const float* Wn1  = (const float*)d_in[7];
    const float* Wo1  = (const float*)d_in[8];
    const float* b1   = (const float*)d_in[9];
    const float* Wn2  = (const float*)d_in[10];
    const float* Wo2  = (const float*)d_in[11];
    const float* b2   = (const float*)d_in[12];
    float* out = (float*)d_out;

    const int N = N_NODES, E = N_EDGES;
    const long NF = (long)N * 128;

    char* ws = (char*)d_ws;
    size_t off = 0;
    auto carve = [&](size_t bytes) {
        char* p = ws + off;
        off = (off + bytes + 255) & ~(size_t)255;
        return p;
    };
    bf16_t* B0   = (bf16_t*)carve(NF * sizeof(bf16_t));
    bf16_t* B1   = (bf16_t*)carve(NF * sizeof(bf16_t));
    float*  hn   = (float*) carve(NF * sizeof(float));
    float*  agg  = (float*) carve(NF * sizeof(float));
    bf16_t* Wn0p = (bf16_t*)carve(128 * 128 * sizeof(bf16_t));
    bf16_t* Wo0p = (bf16_t*)carve(128 * 128 * sizeof(bf16_t));
    bf16_t* Wn1p = (bf16_t*)carve(128 * 128 * sizeof(bf16_t));
    bf16_t* Wo1p = (bf16_t*)carve(128 * 128 * sizeof(bf16_t));
    bf16_t* Wn2p = (bf16_t*)carve(128 * 64  * sizeof(bf16_t));
    bf16_t* Wo2p = (bf16_t*)carve(128 * 64  * sizeof(bf16_t));
    (void)ws_size; (void)n_in; (void)in_sizes; (void)out_size;

    // ---- prep ---------------------------------------------------------------
    {
        int t128 = 128 * 128, t64 = 128 * 64;
        pack_w_kernel<<<(t128 + 255) / 256, 256, 0, stream>>>(Wn0, Wn0p, 128, 128);
        pack_w_kernel<<<(t128 + 255) / 256, 256, 0, stream>>>(Wo0, Wo0p, 128, 128);
        pack_w_kernel<<<(t128 + 255) / 256, 256, 0, stream>>>(Wn1, Wn1p, 128, 128);
        pack_w_kernel<<<(t128 + 255) / 256, 256, 0, stream>>>(Wo1, Wo1p, 128, 128);
        pack_w_kernel<<<(t64  + 255) / 256, 256, 0, stream>>>(Wn2, Wn2p, 128, 64);
        pack_w_kernel<<<(t64  + 255) / 256, 256, 0, stream>>>(Wo2, Wo2p, 128, 64);
        f32_to_bf16_kernel<<<(unsigned)((NF + 255) / 256), 256, 0, stream>>>(x, B0, NF);
    }

    const int rowTiles = N / 16;                           // 6250
    const unsigned gemmGrid = 1024;                        // persistent blocks
    const unsigned sBlocks = (unsigned)((E + 255) / 256);  // 1 thread : 1 edge

    // ---- layer 0 ------------------------------------------------------------
    gemm_dual_kernel<128, 8><<<gemmGrid, 256, 0, stream>>>(B0, Wn0p, Wo0p, b0,
                                                           hn, agg, rowTiles);
    spmm_scatter_kernel<128><<<sBlocks, 256, 0, stream>>>(rows, cols, vals, hn, agg, E);
    tanh_to_bf16_kernel<<<(unsigned)((NF + 255) / 256), 256, 0, stream>>>(agg, B1, NF);

    // ---- layer 1 ------------------------------------------------------------
    gemm_dual_kernel<128, 8><<<gemmGrid, 256, 0, stream>>>(B1, Wn1p, Wo1p, b1,
                                                           hn, agg, rowTiles);
    spmm_scatter_kernel<128><<<sBlocks, 256, 0, stream>>>(rows, cols, vals, hn, agg, E);
    tanh_to_bf16_kernel<<<(unsigned)((NF + 255) / 256), 256, 0, stream>>>(agg, B0, NF);

    // ---- layer 2 (no activation, accumulate straight into d_out) -----------
    gemm_dual_kernel<128, 4><<<gemmGrid, 128, 0, stream>>>(B0, Wn2p, Wo2p, b2,
                                                           hn, out, rowTiles);
    spmm_scatter_kernel<64><<<sBlocks, 256, 0, stream>>>(rows, cols, vals, hn, out, E);
}
